// LSTM_43379169689829
// MI455X (gfx1250) — matrix operand order repaired and checked by reading, not compiled, not run on hardware
//
#include <hip/hip_runtime.h>
#include <stdint.h>

// ---------------------------------------------------------------------------
// LayerNorm-LSTM for MI455X (gfx1250, wave32).
//   Phase 1 (per layer): Gates = LN(x @ Wx)*g+b + bias   -- WMMA f16 GEMM,
//       fused LN epilogue, async-to-LDS double buffering (320KB WGP LDS).
//   Phase 2 (per layer): persistent 32-WG scan kernel, wh slice LDS-resident,
//       WMMA per step, device-wide barriers for the h / LN dependencies.
//   512-thread WGs (16 waves) keep VGPRs < 256; fragment loads are software
//   pipelined so ds latency overlaps WMMA issue (counted s_wait_dscnt).
// ---------------------------------------------------------------------------

typedef __attribute__((ext_vector_type(16))) _Float16 v16h;
typedef __attribute__((ext_vector_type(8)))  _Float16 v8h;
typedef __attribute__((ext_vector_type(8)))  float    v8f;

#define S_SEQ 512
#define BATCH 64
#define HID   512
#define GATE  2048          // 4H
#define MROWS (S_SEQ*BATCH) // 32768
#define NWG   32            // workgroups in the scan kernel

union HF16 { v16h v; v8h h[2]; };

// ---- CDNA5 async copy global->LDS (ASYNCcnt tracked) ----------------------
__device__ __forceinline__ void async_ld16(uint32_t lds, const void* g) {
  asm volatile("global_load_async_to_lds_b128 %0, %1, off"
               :: "v"(lds), "v"((uint64_t)(uintptr_t)g) : "memory");
}
template<int N>
__device__ __forceinline__ void wait_async() {
  asm volatile("s_wait_asynccnt %0" :: "n"(N) : "memory");
}
__device__ __forceinline__ uint32_t lds_addr(const void* p) {
  // flat LDS-aperture address: low 32 bits are the LDS byte offset
  return (uint32_t)(uintptr_t)p;
}

__device__ __forceinline__ float sigmoidf_(float x) {
  return 1.f / (1.f + __expf(-x));
}
__device__ __forceinline__ float tanhf_(float x) {
  float a = fabsf(x);
  float e = __expf(-2.f * a);
  float t = (1.f - e) / (1.f + e);
  return copysignf(t, x);
}

// ===========================================================================
// Kernel 1: C = A[M,512] @ BT[2048,512]^T, fused LayerNorm over the 2048-wide
// row + affine + bias.  One WG = 16 rows x 2048 cols, K streamed in 32-chunks.
// LDS: A tile 16x520h (16.6KB) + 2x B chunk 2048x64B (256KB, rotate-swizzled).
// 16 waves, each owns 128 columns (8 C tiles -> 64 acc VGPRs).
// ===========================================================================
#define GA_BYTES (16*520*2)            // 16640
#define GB_BYTES (GATE*64)             // 131072 per buffer
#define GSM_RED  (GA_BYTES + 2*GB_BYTES)
#define GSM_SIZE (GSM_RED + 16*32*2*4 + 16*2*4)

__global__ __launch_bounds__(512, 1) void k_gemm_ln(
    const _Float16* __restrict__ A, const _Float16* __restrict__ BT,
    const float* __restrict__ gx, const float* __restrict__ bx,
    const float* __restrict__ bias, float* __restrict__ Gates)
{
  extern __shared__ char smem[];
  _Float16* As = (_Float16*)smem;              // [16][520] halves
  char* B0 = smem + GA_BYTES;
  char* B1 = B0 + GB_BYTES;
  float* red = (float*)(smem + GSM_RED);       // [16][32][2]
  float* mv  = red + 16*32*2;                  // [16][2]
  float* Ps  = (float*)B0;                     // reuse B region, [16][2064]

  const int tid = threadIdx.x;
  const int m0  = blockIdx.x * 16;

  // A tile (16 rows x 512 halves), row stride 1040B
#pragma unroll
  for (int it = 0; it < 2; ++it) {
    int u = tid + it*512;
    int r = u >> 6, p = u & 63;
    async_ld16(lds_addr(As) + (uint32_t)(r*1040 + p*16),
               A + (size_t)(m0 + r)*HID + p*8);
  }
  // B chunk 0: col n gets 64B at n*64, 16B parts rotated by (n>>2)&3 -> bank-free
#pragma unroll
  for (int it = 0; it < 16; ++it) {
    int u = tid + it*512;
    int n = u >> 2, p = u & 3;
    async_ld16(lds_addr(B0) + (uint32_t)(n*64 + (((p + (n>>2)) & 3) << 4)),
               BT + (size_t)n*HID + p*8);
  }

  const int wv = tid >> 5, lane = tid & 31;
  const int lr = lane & 15, hh = lane >> 4;
  const int colbase = wv * 128;                // 16 waves x 128 cols

  v8f zero = {0.f,0.f,0.f,0.f,0.f,0.f,0.f,0.f};
  v8f acc[8];
#pragma unroll
  for (int j = 0; j < 8; ++j) acc[j] = zero;

  for (int kc = 0; kc < 16; ++kc) {
    char* Bcur = (kc & 1) ? B1 : B0;
    if (kc + 1 < 16) {                          // prefetch next K-chunk
      char* Bnext = (kc & 1) ? B0 : B1;
      const _Float16* src = BT + (kc+1)*32;
#pragma unroll
      for (int it = 0; it < 16; ++it) {
        int u = tid + it*512;
        int n = u >> 2, p = u & 3;
        async_ld16(lds_addr(Bnext) + (uint32_t)(n*64 + (((p + (n>>2)) & 3) << 4)),
                   src + (size_t)n*HID + p*8);
      }
      wait_async<16>();                         // current chunk (+A) landed
    } else {
      wait_async<0>();
    }
    __syncthreads();

    // A fragment: lane lr=row, lo-run K=kk+16*hh, hi-run K=kk+16*hh+8
    HF16 af;
    {
      const char* ap = (const char*)As + lr*1040 + kc*64 + hh*32;
      af.h[0] = *(const v8h*)ap;
      af.h[1] = *(const v8h*)(ap + 16);
    }
    // software-pipelined B fragments: load j+1 while wmma j executes
    HF16 bf;
    {
      int n = colbase + lr;
      bf.h[0] = *(const v8h*)(Bcur + n*64 + ((((hh<<1)     + (n>>2)) & 3) << 4));
      bf.h[1] = *(const v8h*)(Bcur + n*64 + ((((hh<<1) + 1 + (n>>2)) & 3) << 4));
    }
#pragma unroll
    for (int j = 0; j < 8; ++j) {
      HF16 bfn;
      if (j < 7) {
        int n = colbase + (j+1)*16 + lr;
        bfn.h[0] = *(const v8h*)(Bcur + n*64 + ((((hh<<1)     + (n>>2)) & 3) << 4));
        bfn.h[1] = *(const v8h*)(Bcur + n*64 + ((((hh<<1) + 1 + (n>>2)) & 3) << 4));
      }
      acc[j] = __builtin_amdgcn_wmma_f32_16x16x32_f16(
                 false, af.v, false, bf.v, (short)0, acc[j], false, false);
      if (j < 7) bf = bfn;
    }
    __syncthreads();
  }

  // accumulators -> LDS (C layout: VGPR i = row i or i+8)
#pragma unroll
  for (int j = 0; j < 8; ++j)
#pragma unroll
    for (int i = 0; i < 8; ++i) {
      int row = i + (hh << 3);
      Ps[row*2064 + colbase + j*16 + lr] = acc[j][i];
    }
  __syncthreads();

  // fused LayerNorm over the 2048-wide row
  {
    int rr = tid >> 5, seg = tid & 31;
    float s = 0.f, s2 = 0.f;
    const float* pr = Ps + rr*2064 + seg*64;
    for (int c2 = 0; c2 < 64; ++c2) { float v = pr[c2]; s += v; s2 += v*v; }
    red[(rr*32+seg)*2]   = s;
    red[(rr*32+seg)*2+1] = s2;
  }
  __syncthreads();
  if (tid < 16) {
    float s = 0.f, s2 = 0.f;
    for (int q = 0; q < 32; ++q) { s += red[(tid*32+q)*2]; s2 += red[(tid*32+q)*2+1]; }
    float m   = s * (1.f/GATE);
    float var = s2 * (1.f/GATE) - m*m;
    mv[tid*2]   = m;
    mv[tid*2+1] = rsqrtf(var + 1e-5f);
  }
  __syncthreads();
#pragma unroll 4
  for (int e = 0; e < 64; ++e) {
    int idx = e*512 + tid;
    int r = idx >> 11, c = idx & 2047;
    float v = (Ps[r*2064 + c] - mv[r*2]) * mv[r*2+1] * gx[c] + bx[c] + bias[c];
    Gates[(size_t)(m0 + r)*GATE + c] = v;
  }
}

// ===========================================================================
// Kernel 2: persistent scan.  32 WGs; WG j owns h-cols [16j,16j+16) i.e. gate
// cols {16j..}+{512,1024,1536}.  wh slice (64x512 f16) LDS-resident.
// 16 waves, one 16x16 C tile each; next-kc fragments prefetched.
// ===========================================================================
#define SB_BYTES (64*520*2)     // 66560, wh slice
#define SH_BYTES (64*520*2)     // 66560, h staging
#define SP_FLOATS (64*68)       // P [64][68]
#define SSM_SIZE (SB_BYTES + SH_BYTES + SP_FLOATS*4 + 64*16*4 + 64*8*2*4 + 64*2*4)

__device__ __forceinline__ void grid_barrier(volatile unsigned* cnt, unsigned& epoch) {
  __threadfence();
  __syncthreads();
  epoch += NWG;
  if (threadIdx.x == 0) {
    atomicAdd((unsigned*)cnt, 1u);
    while (*cnt < epoch) __builtin_amdgcn_s_sleep(2);
  }
  __syncthreads();
  __threadfence();
}

__global__ __launch_bounds__(512, 1) void k_scan(
    const float* __restrict__ Gx, const _Float16* __restrict__ whT,
    const float* __restrict__ h0, const float* __restrict__ c0,
    const float* __restrict__ gh, const float* __restrict__ bh,
    _Float16* __restrict__ hbuf, float* __restrict__ pbuf,
    unsigned* __restrict__ syncv,
    _Float16* __restrict__ xnext, float* __restrict__ hsout,
    float* __restrict__ hout, float* __restrict__ cout)
{
  extern __shared__ char smem[];
  char*  Bs  = smem;                        // wh slice [64 cols][520h]
  char*  Hs  = Bs + SB_BYTES;               // h f16     [64 rows][520h]
  float* Ps  = (float*)(Hs + SH_BYTES);     // [64][68]
  float* Cs  = Ps + SP_FLOATS;              // [64][16] cell state (f32)
  float* red8= Cs + 64*16;                  // [64][8][2]
  float* mv  = red8 + 64*8*2;               // [64][2]

  const int tid = threadIdx.x;
  const int wg  = blockIdx.x;
  const int j0  = wg * 16;
  unsigned epoch = 0;
  volatile unsigned* cnt = syncv;

  // resident wh slice: rows rr -> gate col (rr>>4)*512 + j0 + (rr&15)
#pragma unroll
  for (int it = 0; it < 8; ++it) {
    int u = tid + it*512;
    int rr = u >> 6, p = u & 63;
    int gc = (rr >> 4)*HID + j0 + (rr & 15);
    async_ld16(lds_addr(Bs) + (uint32_t)(rr*1040 + p*16),
               whT + (size_t)gc*HID + p*8);
  }
  // init cell state + shared f16 h
#pragma unroll
  for (int e = 0; e < 2; ++e) {
    int idx = tid + e*512;
    int b = idx >> 4, jl = idx & 15;
    Cs[b*16 + jl] = c0[(size_t)b*HID + j0 + jl];
    hbuf[(size_t)b*HID + j0 + jl] = (_Float16)h0[(size_t)b*HID + j0 + jl];
  }
  wait_async<0>();
  __syncthreads();
  grid_barrier(cnt, epoch);

  const int wv = tid >> 5, lane = tid & 31;
  const int lr = lane & 15, hh = lane >> 4;
  const int mt = wv & 3, nt = wv >> 2;      // 4 M-tiles x 4 N-tiles, 1/wave
  v8f zero = {0.f,0.f,0.f,0.f,0.f,0.f,0.f,0.f};
  const char* apB = Hs + (mt*16 + lr)*1040 + hh*16;
  const char* bpB = Bs + (nt*16 + lr)*1040 + hh*16;

  for (int t = 0; t < S_SEQ; ++t) {
    // stage h (64x512 f16) from global into LDS
#pragma unroll
    for (int it = 0; it < 8; ++it) {
      int u = tid + it*512;
      int b = u >> 6, p = u & 63;
      async_ld16(lds_addr(Hs) + (uint32_t)(b*1040 + p*16),
                 hbuf + (size_t)b*HID + p*8);
    }
    wait_async<0>();
    __syncthreads();

    // P tile = h(16 rows) @ wh_slice(16 cols); prefetch kc+1 fragments
    v8f acc = zero;
    HF16 af, bf;
    af.h[0] = *(const v8h*)apB;       af.h[1] = *(const v8h*)(apB + 32);
    bf.h[0] = *(const v8h*)bpB;       bf.h[1] = *(const v8h*)(bpB + 32);
#pragma unroll
    for (int kc = 0; kc < 16; ++kc) {
      HF16 afn, bfn;
      if (kc < 15) {
        const char* ap = apB + (kc+1)*64;
        const char* bp = bpB + (kc+1)*64;
        afn.h[0] = *(const v8h*)ap;   afn.h[1] = *(const v8h*)(ap + 32);
        bfn.h[0] = *(const v8h*)bp;   bfn.h[1] = *(const v8h*)(bp + 32);
      }
      acc = __builtin_amdgcn_wmma_f32_16x16x32_f16(
              false, af.v, false, bf.v, (short)0, acc, false, false);
      if (kc < 15) { af = afn; bf = bfn; }
    }
#pragma unroll
    for (int i = 0; i < 8; ++i) {
      int b = mt*16 + i + (hh << 3);
      Ps[b*68 + nt*16 + lr] = acc[i];
    }
    __syncthreads();

    // local LN partials over this WG's 64 columns
    {
      int b = tid >> 3, q = tid & 7;
      float s = 0.f, s2 = 0.f;
      const float* pr = Ps + b*68 + q*8;
#pragma unroll
      for (int c2 = 0; c2 < 8; ++c2) { float v = pr[c2]; s += v; s2 += v*v; }
      red8[(b*8+q)*2]   = s;
      red8[(b*8+q)*2+1] = s2;
    }
    __syncthreads();
    if (tid < 64) {
      float s = 0.f, s2 = 0.f;
#pragma unroll
      for (int q = 0; q < 8; ++q) { s += red8[(tid*8+q)*2]; s2 += red8[(tid*8+q)*2+1]; }
      pbuf[(wg*BATCH + tid)*2]   = s;
      pbuf[(wg*BATCH + tid)*2+1] = s2;
    }
    grid_barrier(cnt, epoch);

    // global mean / rsqrt(var) over all 2048 columns
    if (tid < 64) {
      float s = 0.f, s2 = 0.f;
      const volatile float* pv = (const volatile float*)pbuf;
      for (int w2 = 0; w2 < NWG; ++w2) {
        s  += pv[(w2*BATCH + tid)*2];
        s2 += pv[(w2*BATCH + tid)*2+1];
      }
      float m   = s * (1.f/GATE);
      float var = s2 * (1.f/GATE) - m*m;
      mv[tid*2]   = m;
      mv[tid*2+1] = rsqrtf(var + 1e-5f);
    }
    __syncthreads();

    // gates + state update for the owned 16 h-columns
#pragma unroll
    for (int e = 0; e < 2; ++e) {
      int idx = tid + e*512;
      int b = idx >> 4, jl = idx & 15;
      float m = mv[b*2], ri = mv[b*2+1];
      const float* pr = Ps + b*68 + jl;
      size_t gbase = ((size_t)t*BATCH + b)*GATE;
      int ci = j0 + jl, cf = 512 + ci, co = 1024 + ci, cu = 1536 + ci;
      float gi = Gx[gbase + ci] + (pr[0]  - m)*ri*gh[ci] + bh[ci];
      float gf = Gx[gbase + cf] + (pr[16] - m)*ri*gh[cf] + bh[cf];
      float go = Gx[gbase + co] + (pr[32] - m)*ri*gh[co] + bh[co];
      float gu = Gx[gbase + cu] + (pr[48] - m)*ri*gh[cu] + bh[cu];
      float c = sigmoidf_(gf)*Cs[b*16+jl] + sigmoidf_(gi)*tanhf_(gu);
      Cs[b*16+jl] = c;
      float h = sigmoidf_(go)*tanhf_(c);
      hbuf[(size_t)b*HID + j0 + jl] = (_Float16)h;
      if (xnext) xnext[((size_t)t*BATCH + b)*HID + j0 + jl] = (_Float16)h;
      if (hsout) hsout[((size_t)t*BATCH + b)*HID + j0 + jl] = h;
      if (t == S_SEQ-1) hout[(size_t)b*HID + j0 + jl] = h;
    }
    grid_barrier(cnt, epoch);
  }

#pragma unroll
  for (int e = 0; e < 2; ++e) {
    int idx = tid + e*512;
    int b = idx >> 4, jl = idx & 15;
    cout[(size_t)b*HID + j0 + jl] = Cs[b*16 + jl];
  }
}

// ===========================================================================
// Helpers: f32 -> f16 convert, and [512,2048] f32 -> [2048,512] f16 transpose
// ===========================================================================
__global__ __launch_bounds__(256) void k_cvt_f16(
    const float* __restrict__ in, _Float16* __restrict__ out, int n) {
  int i = blockIdx.x*256 + threadIdx.x;
  if (i < n) out[i] = (_Float16)in[i];
}

__global__ __launch_bounds__(256) void k_transpose_w(
    const float* __restrict__ in, _Float16* __restrict__ out) {
  __shared__ float tile[32][33];
  int c0b = blockIdx.x * 32;   // gate col
  int r0  = blockIdx.y * 32;   // hid row
  int lx = threadIdx.x & 31, ly = threadIdx.x >> 5;
#pragma unroll
  for (int rr = ly; rr < 32; rr += 8)
    tile[rr][lx] = in[(size_t)(r0+rr)*GATE + c0b + lx];
  __syncthreads();
#pragma unroll
  for (int rr = ly; rr < 32; rr += 8)
    out[(size_t)(c0b+rr)*HID + r0 + lx] = (_Float16)tile[lx][rr];
}

// ===========================================================================
extern "C" void kernel_launch(void* const* d_in, const int* in_sizes, int n_in,
                              void* d_out, int out_size, void* d_ws, size_t ws_size,
                              hipStream_t stream) {
  (void)in_sizes; (void)n_in; (void)out_size; (void)ws_size;
  const float* x    = (const float*)d_in[0];
  const float* h0   = (const float*)d_in[1];
  const float* c0   = (const float*)d_in[2];
  const float* wx   = (const float*)d_in[3];
  const float* wh   = (const float*)d_in[4];
  const float* bias = (const float*)d_in[5];
  const float* lng  = (const float*)d_in[6];
  const float* lnb  = (const float*)d_in[7];

  float* out_x = (float*)d_out;                      // [S,B,H]
  float* out_h = out_x + (size_t)MROWS*HID;          // [2,B,H]
  float* out_c = out_h + (size_t)2*BATCH*HID;        // [2,B,H]

  // workspace carve (needs ~306 MB)
  char* w = (char*)d_ws;
  float*    Gates = (float*)w;      w += (size_t)MROWS*GATE*4;   // 256 MB
  _Float16* xf16  = (_Float16*)w;   w += (size_t)MROWS*HID*2;    // 32 MB
  _Float16* wxT   = (_Float16*)w;   w += (size_t)GATE*HID*2;     // 2 MB
  _Float16* whT   = (_Float16*)w;   w += (size_t)GATE*HID*2;     // 2 MB
  _Float16* hbuf  = (_Float16*)w;   w += (size_t)BATCH*HID*2;    // 64 KB
  float*    pbuf  = (float*)w;      w += (size_t)NWG*BATCH*2*4;  // 16 KB
  unsigned* syncv = (unsigned*)w;   w += 256;

  k_cvt_f16<<<(MROWS*HID + 255)/256, 256, 0, stream>>>(x, xf16, MROWS*HID);

  for (int l = 0; l < 2; ++l) {
    dim3 tg(GATE/32, HID/32);
    k_transpose_w<<<tg, 256, 0, stream>>>(wx + (size_t)l*HID*GATE, wxT);
    k_transpose_w<<<tg, 256, 0, stream>>>(wh + (size_t)l*HID*GATE, whT);

    k_gemm_ln<<<MROWS/16, 512, GSM_SIZE, stream>>>(
        xf16, wxT,
        lng + (size_t)(l*2 + 0)*GATE, lnb + (size_t)(l*2 + 0)*GATE,
        bias + (size_t)l*GATE, Gates);

    hipMemsetAsync(syncv, 0, 256, stream);   // barrier counter -> 0 (capture-safe)

    k_scan<<<NWG, 512, SSM_SIZE, stream>>>(
        Gates, whT,
        h0 + (size_t)l*BATCH*HID, c0 + (size_t)l*BATCH*HID,
        lng + (size_t)(l*2 + 1)*GATE, lnb + (size_t)(l*2 + 1)*GATE,
        hbuf, pbuf, syncv,
        (l == 0) ? xf16 : (_Float16*)nullptr,   // layer0 h feeds layer1 GEMM
        (l == 1) ? out_x : (float*)nullptr,     // layer1 h is the x output
        out_h + (size_t)l*BATCH*HID, out_c + (size_t)l*BATCH*HID);
  }
}
